// LSNN_87050397155446
// MI455X (gfx1250) — matrix operand order repaired and checked by reading, not compiled, not run on hardware
//
#include <hip/hip_runtime.h>
#include <hip/hip_bf16.h>

// LSNN adaptive-LIF spiking RNN for MI455X (gfx1250, wave32, WMMA).
// Dominant op per step: spikes[2048x800] @ W_h2h^T[800x800]  (bf16 WMMA, f32 acc)
// Spikes are exactly {0,1} -> bf16 representation is exact.
// All operands pre-packed into v_wmma_f32_16x16x32_bf16 fragment layout.

typedef __bf16 bf16_t;
typedef __attribute__((ext_vector_type(16))) __bf16 v16bf;
typedef __attribute__((ext_vector_type(8)))  float  v8f;

#define NBATCH 2048
#define HID    800
#define OUTN   10
#define INPUT  28
#define TSTEPS 78
#define TENC   28
#define KT_H   (HID / 32)    // 25 K-tiles over the recurrent dimension
#define NT_H   (HID / 16)    // 50 N-tiles over hidden neurons
#define MT     (NBATCH / 16) // 128 M-tiles over batch

// A-fragment (16x32 bf16) K index for lane-group g (lane/16) and element j:
// group0: j0..7 -> K0..7,  j8..15 -> K16..23 ; group1: j0..7 -> K8..15, j8..15 -> K24..31
__device__ __host__ __forceinline__ int a_kk(int g, int j) {
  return (j & 7) + (g << 3) + ((j >> 3) << 4);
}

// ---------------------------------------------------------------------------
// State / packed-buffer init (runs every call; harness poisons buffers)
// ---------------------------------------------------------------------------
__global__ void lsnn_init(float* h_mem, float* b_h, float* spk_f32, bf16_t* spk_pack,
                          float* o_mem, float* o_b, float* o_spk, float* acc_out) {
  const size_t i = (size_t)blockIdx.x * blockDim.x + threadIdx.x;
  const size_t nh = (size_t)NBATCH * HID;
  if (i < nh) {
    h_mem[i]   = 0.0f;
    b_h[i]     = 0.01f;       // b_j0
    spk_f32[i] = 0.0f;
    spk_pack[i] = (bf16_t)0.0f;
  }
  if (i < (size_t)NBATCH * OUTN) {
    o_mem[i]   = 0.0f;
    o_b[i]     = 0.01f;
    o_spk[i]   = 0.0f;
    acc_out[i] = 0.0f;        // d_out accumulator
  }
}

// ---------------------------------------------------------------------------
// Pack W_i2h (K-tile 0, K padded 28->32) + W_h2h (K-tiles 1..25) into
// B-fragment layout: lane L holds N = nt*16 + L%16, K = kbase + a_kk(L/16, j)
// ---------------------------------------------------------------------------
__global__ void lsnn_prep_B(const float* __restrict__ W_i2h,
                            const float* __restrict__ W_h2h,
                            bf16_t* __restrict__ Bpack) {
  const size_t id = (size_t)blockIdx.x * blockDim.x + threadIdx.x;
  if (id >= (size_t)(KT_H + 1) * NT_H * 32 * 16) return;
  const int j  = id & 15;
  const int L  = (id >> 4) & 31;
  const int nt = (id >> 9) % NT_H;
  const int kt = id / (512 * NT_H);
  const int n  = nt * 16 + (L & 15);
  const int kk = a_kk(L >> 4, j);       // B-fragment K mapping
  float v;
  if (kt == 0) {
    v = (kk < INPUT) ? W_i2h[(size_t)n * INPUT + kk] : 0.0f;
  } else {
    v = W_h2h[(size_t)n * HID + ((kt - 1) * 32 + kk)];
  }
  Bpack[id] = (bf16_t)v;
}

// W_h2o packed (single N-tile, N padded 10->16), 25 K-tiles.
__global__ void lsnn_prep_Bo(const float* __restrict__ W_h2o,
                             bf16_t* __restrict__ Bopack) {
  const size_t id = (size_t)blockIdx.x * blockDim.x + threadIdx.x;
  if (id >= (size_t)KT_H * 32 * 16) return;
  const int j  = id & 15;
  const int L  = (id >> 4) & 31;
  const int kt = id >> 9;
  const int n  = L & 15;
  const int kk = a_kk(L >> 4, j);
  Bopack[id] = (bf16_t)((n < OUTN) ? W_h2o[(size_t)n * HID + (kt * 32 + kk)] : 0.0f);
}

// Input sequence packed into A-fragment layout for the 28 encode steps:
// seq[t][m][i] = x[m, i*28 + t];  K-tile of 32 (features 28..31 = 0).
__global__ void lsnn_prep_X(const float* __restrict__ x,
                            bf16_t* __restrict__ Xpack) {
  const size_t id = (size_t)blockIdx.x * blockDim.x + threadIdx.x;
  if (id >= (size_t)TENC * MT * 32 * 16) return;
  const int j  = id & 15;
  const int L  = (id >> 4) & 31;
  const int mt = (id >> 9) % MT;
  const int t  = id / (512 * MT);
  const int m  = mt * 16 + (L & 15);
  const int kk = a_kk(L >> 4, j);       // A-fragment K mapping
  Xpack[id] = (bf16_t)((kk < INPUT) ? x[(size_t)m * 784 + kk * INPUT + t] : 0.0f);
}

// ---------------------------------------------------------------------------
// Fused hidden step: h_in = spk@W_h2h^T + x_t@W_i2h^T + biases, then
// adaptive-LIF update, writing new spikes (f32 state copy + packed bf16
// A-fragments for the next step / output GEMM).
// Block = 128 threads = 4 waves; wave owns a 16-row M stripe and 2 N-tiles.
// Grid = (32, 25) -> 800 workgroups.
// ---------------------------------------------------------------------------
__global__ __launch_bounds__(128)
void lsnn_hidden_step(const bf16_t* __restrict__ spk_pack_cur,
                      const float*  __restrict__ spk_f32_cur,
                      bf16_t*       __restrict__ spk_pack_nxt,
                      float*        __restrict__ spk_f32_nxt,
                      const bf16_t* __restrict__ Bpack,
                      const bf16_t* __restrict__ Xpack_t,
                      float*        __restrict__ h_mem,
                      float*        __restrict__ b_h,
                      const float*  __restrict__ bias_i2h,
                      const float*  __restrict__ bias_h2h,
                      const float*  __restrict__ tau_adp_h,
                      int has_x) {
  const int tid   = threadIdx.x;
  const int w     = tid >> 5;
  const int L     = tid & 31;
  const int mtile = blockIdx.x * 4 + w;        // 0..127
  const int nt0   = blockIdx.y * 2;            // 2 N-tiles per wave

  const v8f vzero = {0.f, 0.f, 0.f, 0.f, 0.f, 0.f, 0.f, 0.f};
  v8f acc[2] = {vzero, vzero};

  // K-tile 0: input drive (only during the 28 encode steps; uniform branch)
  if (has_x) {
    v16bf a = *(const v16bf*)(Xpack_t + (((size_t)mtile) * 32 + L) * 16);
#pragma unroll
    for (int nt = 0; nt < 2; ++nt) {
      v16bf b = *(const v16bf*)(Bpack + (((size_t)0 * NT_H + (nt0 + nt)) * 32 + L) * 16);
      acc[nt] = __builtin_amdgcn_wmma_f32_16x16x32_bf16(
          false, a, false, b, (short)0, acc[nt], false, false);
    }
  }

  // K-tiles 1..25: recurrent drive from current spikes
  for (int kt = 0; kt < KT_H; ++kt) {
    const bf16_t* ap = spk_pack_cur + (((size_t)mtile * KT_H + kt) * 32 + L) * 16;
    if (kt + 1 < KT_H)
      __builtin_prefetch(spk_pack_cur + (((size_t)mtile * KT_H + kt + 1) * 32 + L) * 16, 0, 3);
    v16bf a = *(const v16bf*)ap;
#pragma unroll
    for (int nt = 0; nt < 2; ++nt) {
      v16bf b = *(const v16bf*)(Bpack + (((size_t)(kt + 1) * NT_H + (nt0 + nt)) * 32 + L) * 16);
      acc[nt] = __builtin_amdgcn_wmma_f32_16x16x32_bf16(
          false, a, false, b, (short)0, acc[nt], false, false);
    }
  }

  // Fused adaptive-LIF update (elementwise in C-fragment layout)
  const float alpha     = 0.951229424500714f;  // exp(-1/20)
  const int   mloc_base = (L >> 4) << 3;       // C: lanes16-31 hold M = r+8
  const int   nl        = L & 15;

#pragma unroll
  for (int nt = 0; nt < 2; ++nt) {
    const int   n    = (nt0 + nt) * 16 + nl;
    const float ro   = __expf(-1.0f / tau_adp_h[n]);
    const float bias = bias_i2h[n] + bias_h2h[n];
#pragma unroll
    for (int r = 0; r < 8; ++r) {
      const int    mloc = r + mloc_base;
      const int    m    = mtile * 16 + mloc;
      const size_t si   = (size_t)m * HID + n;
      const float  hin  = acc[nt][r] + bias;
      const float  sold = spk_f32_cur[si];
      float bh = b_h[si];
      bh = ro * bh + (1.0f - ro) * sold;
      const float Bth = 0.01f + 1.8f * bh;
      float mem = h_mem[si];
      mem = mem * alpha + (1.0f - alpha) * hin - Bth * sold;
      const float snew = (mem - Bth) > 0.0f ? 1.0f : 0.0f;
      h_mem[si] = mem;
      b_h[si]   = bh;
      spk_f32_nxt[si] = snew;
      // scatter into A-fragment layout for next step's WMMA
      const int kt2   = n >> 5;
      const int kk    = n & 31;
      const int grp   = (kk >> 3) & 1;
      const int idx   = (kk & 7) + ((kk >> 4) << 3);
      const int lane2 = mloc + (grp << 4);
      spk_pack_nxt[(((size_t)mtile * KT_H + kt2) * 32 + lane2) * 16 + idx] = (bf16_t)snew;
    }
  }
}

// ---------------------------------------------------------------------------
// Fused output step: o_in = spk@W_h2o^T + b_h2o, adaptive-LIF on 10 output
// neurons, accumulate spikes on decode steps. One N-tile (10 padded to 16).
// Grid = 32 blocks x 128 threads (4 waves).
// ---------------------------------------------------------------------------
__global__ __launch_bounds__(128)
void lsnn_output_step(const bf16_t* __restrict__ spk_pack,
                      const bf16_t* __restrict__ Bopack,
                      float* __restrict__ o_mem,
                      float* __restrict__ o_b,
                      float* __restrict__ o_spk,
                      float* __restrict__ acc_out,
                      const float* __restrict__ bias_h2o,
                      const float* __restrict__ tau_adp_o,
                      int is_decode) {
  const int tid   = threadIdx.x;
  const int w     = tid >> 5;
  const int L     = tid & 31;
  const int mtile = blockIdx.x * 4 + w;

  const v8f vzero = {0.f, 0.f, 0.f, 0.f, 0.f, 0.f, 0.f, 0.f};
  v8f acc = vzero;
  for (int kt = 0; kt < KT_H; ++kt) {
    v16bf a = *(const v16bf*)(spk_pack + (((size_t)mtile * KT_H + kt) * 32 + L) * 16);
    v16bf b = *(const v16bf*)(Bopack + (((size_t)kt) * 32 + L) * 16);
    acc = __builtin_amdgcn_wmma_f32_16x16x32_bf16(
        false, a, false, b, (short)0, acc, false, false);
  }

  const float alpha     = 0.951229424500714f;
  const int   mloc_base = (L >> 4) << 3;
  const int   nl        = L & 15;
  if (nl < OUTN) {
    const float ro   = __expf(-1.0f / tau_adp_o[nl]);
    const float bias = bias_h2o[nl];
    const float dec  = (float)is_decode;
#pragma unroll
    for (int r = 0; r < 8; ++r) {
      const int    m  = mtile * 16 + r + mloc_base;
      const size_t si = (size_t)m * OUTN + nl;
      const float  oin  = acc[r] + bias;
      const float  sold = o_spk[si];
      float ob = o_b[si];
      ob = ro * ob + (1.0f - ro) * sold;
      const float Bo = 0.01f + 1.8f * ob;
      float mem = o_mem[si];
      mem = mem * alpha + (1.0f - alpha) * oin - Bo * sold;
      const float snew = (mem - Bo) > 0.0f ? 1.0f : 0.0f;
      o_mem[si] = mem;
      o_b[si]   = ob;
      o_spk[si] = snew;
      acc_out[si] += dec * snew;
    }
  }
}

// ---------------------------------------------------------------------------
extern "C" void kernel_launch(void* const* d_in, const int* in_sizes, int n_in,
                              void* d_out, int out_size, void* d_ws, size_t ws_size,
                              hipStream_t stream) {
  const float* x      = (const float*)d_in[0];
  const float* W_i2h  = (const float*)d_in[1];
  const float* b_i2h  = (const float*)d_in[2];
  const float* W_h2h  = (const float*)d_in[3];
  const float* b_h2h  = (const float*)d_in[4];
  const float* W_h2o  = (const float*)d_in[5];
  const float* b_h2o  = (const float*)d_in[6];
  const float* tau_h  = (const float*)d_in[7];
  const float* tau_o  = (const float*)d_in[8];
  float* out = (float*)d_out;

  // Workspace carve-out (256B aligned regions)
  char* p = (char*)d_ws;
  auto carve = [&](size_t bytes) -> char* {
    char* r = p;
    p += (bytes + 255) & ~(size_t)255;
    return r;
  };
  const size_t NH = (size_t)NBATCH * HID;          // 1,638,400
  bf16_t* Bpack  = (bf16_t*)carve((size_t)(KT_H + 1) * NT_H * 512 * sizeof(bf16_t));
  bf16_t* Bopack = (bf16_t*)carve((size_t)KT_H * 512 * sizeof(bf16_t));
  bf16_t* Xpack  = (bf16_t*)carve((size_t)TENC * MT * 512 * sizeof(bf16_t));
  bf16_t* SpkP0  = (bf16_t*)carve(NH * sizeof(bf16_t));
  bf16_t* SpkP1  = (bf16_t*)carve(NH * sizeof(bf16_t));
  float*  SpkF0  = (float*)carve(NH * sizeof(float));
  float*  SpkF1  = (float*)carve(NH * sizeof(float));
  float*  Hmem   = (float*)carve(NH * sizeof(float));
  float*  BhS    = (float*)carve(NH * sizeof(float));
  float*  Omem   = (float*)carve((size_t)NBATCH * OUTN * sizeof(float));
  float*  Ob     = (float*)carve((size_t)NBATCH * OUTN * sizeof(float));
  float*  Ospk   = (float*)carve((size_t)NBATCH * OUTN * sizeof(float));

  bf16_t* SpkP[2] = {SpkP0, SpkP1};
  float*  SpkF[2] = {SpkF0, SpkF1};

  // 1) init state + output accumulator
  lsnn_init<<<(int)((NH + 255) / 256), 256, 0, stream>>>(
      Hmem, BhS, SpkF[0], SpkP[0], Omem, Ob, Ospk, out);

  // 2) pack weights + inputs into WMMA fragment layout
  {
    const size_t nB = (size_t)(KT_H + 1) * NT_H * 512;
    lsnn_prep_B<<<(int)((nB + 255) / 256), 256, 0, stream>>>(W_i2h, W_h2h, Bpack);
    const size_t nBo = (size_t)KT_H * 512;
    lsnn_prep_Bo<<<(int)((nBo + 255) / 256), 256, 0, stream>>>(W_h2o, Bopack);
    const size_t nX = (size_t)TENC * MT * 512;
    lsnn_prep_X<<<(int)((nX + 255) / 256), 256, 0, stream>>>(x, Xpack);
  }

  // 3) sequential timestep chain (graph-captured; 2 kernels/step)
  int cur = 0;
  for (int t = 0; t < TSTEPS; ++t) {
    const int nxt = cur ^ 1;
    const int has_x = (t < TENC) ? 1 : 0;
    const bf16_t* Xp = Xpack + (size_t)(has_x ? t : 0) * MT * 512;
    lsnn_hidden_step<<<dim3(MT / 4, NT_H / 2), 128, 0, stream>>>(
        SpkP[cur], SpkF[cur], SpkP[nxt], SpkF[nxt],
        Bpack, Xp, Hmem, BhS, b_i2h, b_h2h, tau_h, has_x);
    lsnn_output_step<<<MT / 4, 128, 0, stream>>>(
        SpkP[nxt], Bopack, Omem, Ob, Ospk, out, b_h2o, tau_o, (t >= TENC) ? 1 : 0);
    cur = nxt;
  }
}
